// DecoderDCRNN_7490422964885
// MI455X (gfx1250) — matrix operand order repaired and checked by reading, not compiled, not run on hardware
//
#include <hip/hip_runtime.h>
#include <hip/hip_bf16.h>

// ---------------------------------------------------------------------------
// DCRNN decoder for MI455X (gfx1250, wave32).
//   - All dense GEMMs via v_wmma_f32_16x16x32_bf16 (fp32 accumulate).
//   - Weights staged in LDS pre-swizzled into WMMA B-fragment order:
//     fragment reads are 2x ds_load_b128 per WMMA, no per-element packing.
//   - Graph propagation: coalesced float4 gathers + global_atomic_add_f32.
// ---------------------------------------------------------------------------

#define LL   2
#define KCH  3
#define HID  64
#define OUTD 32
#define BB   8
#define NN   10000
#define EE   160000
#define FF   128                 // 2*HID
#define BN   (BB * NN)           // 80000 rows
#define WSZ  (FF * HID)          // 8192 elements per (l, dir, k) weight block

typedef float  v8f   __attribute__((ext_vector_type(8)));
typedef __bf16 v16bf __attribute__((ext_vector_type(16)));
typedef __bf16 v2bf  __attribute__((ext_vector_type(2)));

static __device__ __forceinline__ __bf16 f2bf(float x) { return (__bf16)x; }

// ---------------------------- elementwise kernels --------------------------

__global__ void k_zero4(float4* __restrict__ p, int n4) {
    int t = blockIdx.x * blockDim.x + threadIdx.x;
    if (t < n4) p[t] = float4{0.f, 0.f, 0.f, 0.f};
}

// o = a + b (small weight combine: W[0,0] + W[1,0])
__global__ void k_wsum(const float* __restrict__ a, const float* __restrict__ b,
                       float* __restrict__ o, int n) {
    int t = blockIdx.x * blockDim.x + threadIdx.x;
    if (t < n) o[t] = a[t] + b[t];
}

// Chebyshev combine: y = 2*y - x   (float4 vectorized)
__global__ void k_cheb(float4* __restrict__ y, const float4* __restrict__ x, int n4) {
    int t = blockIdx.x * blockDim.x + threadIdx.x;
    if (t >= n4) return;
    float4 v = y[t]; const float4 u = x[t];
    v.x = 2.f * v.x - u.x; v.y = 2.f * v.y - u.y;
    v.z = 2.f * v.z - u.z; v.w = 2.f * v.w - u.w;
    y[t] = v;
}

// xh = concat(x, h) along feature dim (64+64 = 128)
__global__ void k_concat(const float* __restrict__ x, const float* __restrict__ h,
                         float* __restrict__ o) {
    int t = blockIdx.x * blockDim.x + threadIdx.x;   // BN*FF threads
    int bn = t >> 7, f = t & 127;
    o[t] = (f < HID) ? x[bn * HID + f] : h[bn * HID + (f - HID)];
}

// xh = concat(x, sigmoid(racc) * h)   (fuses R-gate activation)
__global__ void k_concat_rh(const float* __restrict__ x, const float* __restrict__ h,
                            const float* __restrict__ racc, float* __restrict__ o) {
    int t = blockIdx.x * blockDim.x + threadIdx.x;
    int bn = t >> 7, f = t & 127;
    if (f < HID) {
        o[t] = x[bn * HID + f];
    } else {
        int i = bn * HID + (f - HID);
        float r = 1.f / (1.f + expf(-racc[i]));
        o[t] = r * h[i];
    }
}

__global__ void k_sigmoid(const float* __restrict__ in, float* __restrict__ o, int n) {
    int t = blockIdx.x * blockDim.x + threadIdx.x;
    if (t < n) o[t] = 1.f / (1.f + expf(-in[t]));
}

// h_new = z*h + (1-z)*tanh(ht)
__global__ void k_hnew(const float* __restrict__ z, const float* __restrict__ h,
                       const float* __restrict__ ht, float* __restrict__ o, int n) {
    int t = blockIdx.x * blockDim.x + threadIdx.x;
    if (t >= n) return;
    float zz = z[t];
    o[t] = zz * h[t] + (1.f - zz) * tanhf(ht[t]);
}

// weighted degree accumulation
__global__ void k_deg(const float* __restrict__ ew, const int* __restrict__ src,
                      const int* __restrict__ dst, float* __restrict__ degO,
                      float* __restrict__ degI) {
    int e = blockIdx.x * blockDim.x + threadIdx.x;
    if (e >= EE) return;
    float w = ew[e];
    atomicAdd(&degO[src[e]], w);
    atomicAdd(&degI[dst[e]], w);
}

// norm_out[e] = w/deg_out[src], norm_in[e] = w/deg_in[dst]  (guard deg==0)
__global__ void k_norm(const float* __restrict__ ew, const int* __restrict__ src,
                       const int* __restrict__ dst, const float* __restrict__ degO,
                       const float* __restrict__ degI, float* __restrict__ no,
                       float* __restrict__ ni) {
    int e = blockIdx.x * blockDim.x + threadIdx.x;
    if (e >= EE) return;
    float w  = ew[e];
    float dO = degO[src[e]], dI = degI[dst[e]];
    no[e] = w * (dO > 0.f ? 1.f / dO : 1.f);
    ni[e] = w * (dI > 0.f ? 1.f / dI : 1.f);
}

// Y[b, scat[e], :] += w[e] * X[b, gath[e], :]
// Grid: EE blocks x 256 threads; thread = (b, f4). Edge indices are
// block-uniform -> scalar loads; gather/scatter are float4-coalesced.
__global__ void k_prop(const float* __restrict__ X, float* __restrict__ Y,
                       const int* __restrict__ gath, const int* __restrict__ scat,
                       const float* __restrict__ w) {
    int e  = blockIdx.x;
    int b  = threadIdx.x >> 5;          // 0..7
    int f4 = (threadIdx.x & 31) << 2;   // 0,4,...,124
    float wgt = w[e];
    int g = gath[e], s = scat[e];
    const float4 v = *(const float4*)(X + ((long)b * NN + g) * FF + f4);
    float* yp = Y + ((long)b * NN + s) * FF + f4;
    atomicAdd(yp + 0, v.x * wgt);
    atomicAdd(yp + 1, v.y * wgt);
    atomicAdd(yp + 2, v.z * wgt);
    atomicAdd(yp + 3, v.w * wgt);
}

// ------------------------------ WMMA GEMM ----------------------------------
// H[M, Nd] (+)= X[M, Kd] @ Wf[Kd, Nd]   (bf16 inputs, fp32 accumulate)
// One wave computes a 16 x Nd output tile: Kd/32 WMMA k-steps x Nd/16 col tiles.
// Weights are staged in LDS in B-fragment order:
//   fragment (kk, c): lane holds bf16 element e at
//       W[kk*32 + (lane>>4)*8 + (e&7) + (e>>3)*16][c*16 + (lane&15)]
//   stored contiguously: 32 bytes per lane, 1 KB per fragment.
// init==1: H = acc + bias (broadcast over rows); init==0: H += acc.
template <int Kd, int Nd>
__global__ void k_gemm_wmma(const float* __restrict__ X, const float* __restrict__ Wf,
                            float* __restrict__ H, const float* __restrict__ bias,
                            int M, int init) {
    constexpr int NG = Nd / 16;                       // column groups
    constexpr int NK = Kd / 32;                       // k-steps
    __shared__ unsigned int Wl[Kd * Nd / 2];          // bf16-pairs, frag order

    const int tid = threadIdx.x;
    // Stage + swizzle: dword D = fragIdx*256 + lane*8 + dw
    for (int D = tid; D < Kd * Nd / 2; D += 256) {
        const int dw   = D & 7;
        const int ln   = (D >> 3) & 31;
        const int fi   = D >> 8;
        const int kk   = fi / NG;
        const int c    = fi % NG;
        const int row  = (kk << 5) + ((ln >> 4) << 3) + (dw << 1) + ((dw >> 2) << 3);
        const int col  = (c << 4) + (ln & 15);
        v2bf p;
        p[0] = f2bf(Wf[row * Nd + col]);
        p[1] = f2bf(Wf[(row + 1) * Nd + col]);
        Wl[D] = __builtin_bit_cast(unsigned int, p);
    }
    __syncthreads();

    const int wave = tid >> 5;
    const int lane = tid & 31;
    const int half = lane >> 4;         // which 16-lane half of the wave
    const int lr   = lane & 15;
    const int row0 = (blockIdx.x * 8 + wave) * 16;
    if (row0 >= M) return;              // wave-uniform: EXEC stays all-ones

    const v8f vz = {0.f, 0.f, 0.f, 0.f, 0.f, 0.f, 0.f, 0.f};
    v8f acc[NG];
#pragma unroll
    for (int c = 0; c < NG; ++c) acc[c] = vz;

#pragma unroll
    for (int kk = 0; kk < NK; ++kk) {
        const int base = kk << 5;
        // A fragment (16x32 bf16, ISA 7.12.2): lane<16 holds K 0..7,16..23 of
        // its row; lane>=16 holds K 8..15,24..31.  4x global_load_b128.
        const float* xr = X + (long)(row0 + lr) * Kd + base;
        const float4 p0 = *(const float4*)(xr + half * 8);
        const float4 p1 = *(const float4*)(xr + half * 8 + 4);
        const float4 p2 = *(const float4*)(xr + 16 + half * 8);
        const float4 p3 = *(const float4*)(xr + 16 + half * 8 + 4);
        v16bf a;
        a[0]  = f2bf(p0.x); a[1]  = f2bf(p0.y); a[2]  = f2bf(p0.z); a[3]  = f2bf(p0.w);
        a[4]  = f2bf(p1.x); a[5]  = f2bf(p1.y); a[6]  = f2bf(p1.z); a[7]  = f2bf(p1.w);
        a[8]  = f2bf(p2.x); a[9]  = f2bf(p2.y); a[10] = f2bf(p2.z); a[11] = f2bf(p2.w);
        a[12] = f2bf(p3.x); a[13] = f2bf(p3.y); a[14] = f2bf(p3.z); a[15] = f2bf(p3.w);
#pragma unroll
        for (int c = 0; c < NG; ++c) {
            // B fragment: contiguous 32 bytes per lane -> 2x ds_load_b128.
            const v16bf bfr = *reinterpret_cast<const v16bf*>(
                &Wl[((kk * NG + c) * 32 + lane) * 8]);
            acc[c] = __builtin_amdgcn_wmma_f32_16x16x32_bf16(
                false, a, false, bfr, (short)0, acc[c], false, false);
        }
    }

    // D layout: VGPR j <-> row (half*8 + j), column = lr within each group.
#pragma unroll
    for (int c = 0; c < NG; ++c) {
        const int col = (c << 4) + lr;
        const float bval = init ? bias[col] : 0.f;
#pragma unroll
        for (int j = 0; j < 8; ++j) {
            const long idx = (long)(row0 + half * 8 + j) * Nd + col;
            if (init) H[idx] = acc[c][j] + bval;
            else      H[idx] += acc[c][j];
        }
    }
}

// ------------------------------ host-side DAG ------------------------------

static void run_prop(const float* X, float* Y, const int* gi, const int* si,
                     const float* w, hipStream_t s) {
    const int n4 = BN * FF / 4;
    k_zero4<<<(n4 + 255) / 256, 256, 0, s>>>((float4*)Y, n4);
    k_prop<<<EE, 256, 0, s>>>(X, Y, gi, si, w);
}

// One diffusion convolution: Hacc = bias + sum_k T_k @ W[dir,k]
static void dconv(const float* xh, const float* Wg, const float* bias,
                  float* Hacc, float* big1, float* big2, float* Wcomb,
                  const int* src, const int* dst, const float* no,
                  const float* ni, hipStream_t s) {
    const int g625 = BN / 128;                       // 625 blocks (8 waves each)
    const int n4   = BN * FF / 4;
    // T0 term with combined weights W[0,0]+W[1,0]
    k_wsum<<<WSZ / 256, 256, 0, s>>>(Wg, Wg + (size_t)KCH * WSZ, Wcomb, WSZ);
    k_gemm_wmma<FF, HID><<<g625, 256, 0, s>>>(xh, Wcomb, Hacc, bias, BN, 1);
    // forward walk (gather src -> scatter dst, norm_out)
    run_prop(xh, big1, src, dst, no, s);
    k_gemm_wmma<FF, HID><<<g625, 256, 0, s>>>(big1, Wg + 1 * WSZ, Hacc, bias, BN, 0);
    run_prop(big1, big2, src, dst, no, s);
    k_cheb<<<(n4 + 255) / 256, 256, 0, s>>>((float4*)big2, (const float4*)xh, n4);
    k_gemm_wmma<FF, HID><<<g625, 256, 0, s>>>(big2, Wg + 2 * WSZ, Hacc, bias, BN, 0);
    // reverse walk (gather dst -> scatter src, norm_in)
    run_prop(xh, big1, dst, src, ni, s);
    k_gemm_wmma<FF, HID><<<g625, 256, 0, s>>>(big1, Wg + (KCH + 1) * WSZ, Hacc, bias, BN, 0);
    run_prop(big1, big2, dst, src, ni, s);
    k_cheb<<<(n4 + 255) / 256, 256, 0, s>>>((float4*)big2, (const float4*)xh, n4);
    k_gemm_wmma<FF, HID><<<g625, 256, 0, s>>>(big2, Wg + (KCH + 2) * WSZ, Hacc, bias, BN, 0);
}

extern "C" void kernel_launch(void* const* d_in, const int* in_sizes, int n_in,
                              void* d_out, int out_size, void* d_ws, size_t ws_size,
                              hipStream_t stream) {
    (void)in_sizes; (void)n_in; (void)out_size; (void)ws_size;
    const float* ew     = (const float*)d_in[0];
    const float* hidden = (const float*)d_in[1];   // [L,B,N,HID]
    const float* go     = (const float*)d_in[2];   // [B,N,HID]
    const float* Wz     = (const float*)d_in[3];   // [L,2,K,128,64]
    const float* bz     = (const float*)d_in[4];
    const float* Wr     = (const float*)d_in[5];
    const float* br     = (const float*)d_in[6];
    const float* Wh     = (const float*)d_in[7];
    const float* bh     = (const float*)d_in[8];
    const float* projW  = (const float*)d_in[9];   // [64,32]
    const float* projb  = (const float*)d_in[10];  // [32]
    const int*   eidx   = (const int*)d_in[11];    // [2,E]
    const int* src = eidx;
    const int* dst = eidx + EE;

    float* ws    = (float*)d_ws;
    float* big0  = ws;                              // xh       [B,N,128]
    float* big1  = big0  + (size_t)BN * FF;         // T_prev
    float* big2  = big1  + (size_t)BN * FF;         // T_cur
    float* Hacc  = big2  + (size_t)BN * FF;         // dconv accumulator [B,N,64]
    float* Zbuf  = Hacc  + (size_t)BN * HID;
    float* cur   = Zbuf  + (size_t)BN * HID;        // layer output / next input
    float* Wcomb = cur   + (size_t)BN * HID;        // [128,64]
    float* normO = Wcomb + WSZ;                     // per-edge out-normalization
    float* normI = normO + EE;                      // per-edge in-normalization
    float* degO  = normI + EE;
    float* degI  = degO  + NN;                      // total ~186 MB of d_ws

    // ---- edge normalization ----
    k_zero4<<<(2 * NN / 4 + 255) / 256, 256, 0, stream>>>((float4*)degO, 2 * NN / 4);
    k_deg <<<(EE + 255) / 256, 256, 0, stream>>>(ew, src, dst, degO, degI);
    k_norm<<<(EE + 255) / 256, 256, 0, stream>>>(ew, src, dst, degO, degI, normO, normI);

    const int gE  = BN * FF / 256;   // 40000 blocks for [B,N,128] elementwise
    const int gH  = BN * HID / 256;  // 20000 blocks for [B,N,64] elementwise
    const size_t wlstride = (size_t)2 * KCH * WSZ;  // per-layer weight stride

    for (int l = 0; l < LL; ++l) {
        const float* x = (l == 0) ? go : cur;
        const float* h = hidden + (size_t)l * BN * HID;
        // Z gate
        k_concat<<<gE, 256, 0, stream>>>(x, h, big0);
        dconv(big0, Wz + l * wlstride, bz + l * HID, Hacc, big1, big2, Wcomb,
              src, dst, normO, normI, stream);
        k_sigmoid<<<gH, 256, 0, stream>>>(Hacc, Zbuf, BN * HID);
        // R gate (activation fused into candidate-input builder)
        dconv(big0, Wr + l * wlstride, br + l * HID, Hacc, big1, big2, Wcomb,
              src, dst, normO, normI, stream);
        k_concat_rh<<<gE, 256, 0, stream>>>(x, h, Hacc, big0);
        // candidate
        dconv(big0, Wh + l * wlstride, bh + l * HID, Hacc, big1, big2, Wcomb,
              src, dst, normO, normI, stream);
        k_hnew<<<gH, 256, 0, stream>>>(Zbuf, h, Hacc, cur, BN * HID);
    }

    // ---- output projection: [80000,64] @ [64,32] + b ----
    k_gemm_wmma<HID, OUTD><<<BN / 128, 256, 0, stream>>>(
        cur, projW, (float*)d_out, projb, BN, 1);
}
